// GRUModel_22256520528126
// MI455X (gfx1250) — compile-verified, matrix operand order checked by reading, not run on hardware
//
#include <hip/hip_runtime.h>
#include <cstdint>

#define T_LEN 16384
#define IN_DIM 512
#define H_DIM 1024
#define G_DIM (3 * H_DIM)

typedef __attribute__((ext_vector_type(16))) __bf16 v16bf;
typedef __attribute__((ext_vector_type(8)))  float  v8f;
typedef __attribute__((ext_vector_type(4)))  unsigned int u32x4;
typedef __attribute__((ext_vector_type(8)))  int i32x8;
typedef __attribute__((ext_vector_type(4)))  int i32x4;

union Frag32 {
  uint4 u[2];
  v16bf v;
};

__device__ __forceinline__ uint16_t f2bf(float f) {
  unsigned u = __float_as_uint(f);
  return (uint16_t)((u + 0x7FFFu + ((u >> 16) & 1u)) >> 16);
}

__device__ __forceinline__ float sigmoidf_(float x) {
  return 1.0f / (1.0f + __expf(-x));
}

#define WMMA_BF16(A, B, Cacc) \
  __builtin_amdgcn_wmma_f32_16x16x32_bf16(false, (A), false, (B), (short)0, \
                                          (Cacc), false, false)

// ---------------------------------------------------------------------------
// f32 -> bf16 conversion (round to nearest even)
// ---------------------------------------------------------------------------
__global__ void f32_to_bf16(const float* __restrict__ src,
                            uint16_t* __restrict__ dst, long n) {
  long i = (long)blockIdx.x * blockDim.x + threadIdx.x;
  if (i < n) dst[i] = f2bf(src[i]);
}

// ---------------------------------------------------------------------------
// C[M,N] = A[M,K](bf16) @ Bt[N,K]^T(bf16) + bias[N], f32 out.
// 32x32 tile per wave: 2x2 register-blocked WMMA (4 accumulators), so each
// 32-deep K-chunk costs 8 x 16B loads for 4 WMMAs (2x the AI of 1 tile/wave).
// A-frag layout (ISA 7.12.2, 16-bit A 16x32): lane L holds row M=(L&15);
// lanes 0-15 hold K = {0..7, 16..23}, lanes 16-31 hold K = {8..15, 24..31}.
// B mirrors A with lane -> column N.
// ---------------------------------------------------------------------------
__global__ void __launch_bounds__(128)
gemm_bias_bf16(const uint16_t* __restrict__ A,
               const uint16_t* __restrict__ Bt,
               const float* __restrict__ bias,
               float* __restrict__ C,
               int M, int N, int K) {
  const int lane = threadIdx.x & 31;
  const int wave = threadIdx.x >> 5;
  const long tile = (long)blockIdx.x * (blockDim.x >> 5) + wave;
  const int ntn = N >> 5;
  const long ntiles = (long)(M >> 5) * ntn;
  if (tile >= ntiles) return;
  const int tn = (int)(tile % ntn);
  const int tm = (int)(tile / ntn);
  const int kb = (lane & 16) ? 8 : 0;
  const int l15 = lane & 15;
  const uint16_t* a0p = A + (long)(tm * 32 + l15) * K + kb;
  const uint16_t* a1p = A + (long)(tm * 32 + 16 + l15) * K + kb;
  const uint16_t* b0p = Bt + (long)(tn * 32 + l15) * K + kb;
  const uint16_t* b1p = Bt + (long)(tn * 32 + 16 + l15) * K + kb;
  v8f acc00 = {}, acc01 = {}, acc10 = {}, acc11 = {};
  for (int k0 = 0; k0 < K; k0 += 32) {
    Frag32 a0, a1, b0, b1;
    a0.u[0] = *(const uint4*)(a0p + k0);
    a0.u[1] = *(const uint4*)(a0p + k0 + 16);
    a1.u[0] = *(const uint4*)(a1p + k0);
    a1.u[1] = *(const uint4*)(a1p + k0 + 16);
    b0.u[0] = *(const uint4*)(b0p + k0);
    b0.u[1] = *(const uint4*)(b0p + k0 + 16);
    b1.u[0] = *(const uint4*)(b1p + k0);
    b1.u[1] = *(const uint4*)(b1p + k0 + 16);
    __builtin_prefetch(a0p + k0 + 64, 0, 3);
    __builtin_prefetch(b0p + k0 + 64, 0, 3);
    acc00 = WMMA_BF16(a0.v, b0.v, acc00);
    acc01 = WMMA_BF16(a0.v, b1.v, acc01);
    acc10 = WMMA_BF16(a1.v, b0.v, acc10);
    acc11 = WMMA_BF16(a1.v, b1.v, acc11);
  }
  const int c0 = tn * 32 + l15;
  const int c1 = c0 + 16;
  const float bs0 = bias[c0];
  const float bs1 = bias[c1];
  // D layout: VGPR v, lanes 0-15 -> row v, lanes 16-31 -> row v+8.
  const int r0 = tm * 32 + ((lane & 16) ? 8 : 0);
  const int r1 = r0 + 16;
#pragma unroll
  for (int v = 0; v < 8; ++v) {
    C[(long)(r0 + v) * N + c0] = acc00[v] + bs0;
    C[(long)(r0 + v) * N + c1] = acc01[v] + bs1;
    C[(long)(r1 + v) * N + c0] = acc10[v] + bs0;
    C[(long)(r1 + v) * N + c1] = acc11[v] + bs1;
  }
}

// ---------------------------------------------------------------------------
// Zero the barrier + hidden-state double buffers before a scan.
// ---------------------------------------------------------------------------
__global__ void scan_init(int* __restrict__ bar, float* __restrict__ hf,
                          uint16_t* __restrict__ hb) {
  int i = blockIdx.x * blockDim.x + threadIdx.x;
  if (i < 2) bar[i] = 0;
  if (i < 2 * H_DIM) { hf[i] = 0.0f; hb[i] = 0; }
}

// ---------------------------------------------------------------------------
// TDM: stage one contiguous 2D weight chunk (rows x 1024 bf16, row stride
// 1024 elements) from global into LDS via the Tensor Data Mover.
// D# per CDNA5 ISA ch.8: group0 = {count=1, lds_addr, global_addr, type=2},
// group1 = {data_size=2B, tensor_dim0=1024, tensor_dim1=rows, tile 1024xrows,
// tensor_dim0_stride=1024}. Groups 2/3 zero (2D tensor).
// This toolchain exposes the 6-arg builtin:
//   (u32x4 g0, i32x8 g1, i32x4 g2, i32x4 g3, i32x8 extra, i32 cpol)
// ---------------------------------------------------------------------------
#if __has_builtin(__builtin_amdgcn_tensor_load_to_lds)
#define HAVE_TDM 1
__device__ __forceinline__ void tdm_load_2d(unsigned lds_addr,
                                            unsigned long long ga, int rows) {
  u32x4 g0 = {1u,                                   // count=1, user mode
              lds_addr,                             // LDS byte address
              (unsigned)(ga & 0xFFFFFFFFu),         // global addr [31:0]
              (unsigned)(ga >> 32) | (2u << 30)};   // [56:32] | type=2
  i32x8 g1 = {(int)(1u << 16),                 // data_size = 1 (2 bytes)
              (int)(1024u << 16),              // tensor_dim0 = 1024 (lo16)
              (int)((unsigned)rows << 16),     // td0 hi=0 | tensor_dim1 lo16
              (int)(1024u << 16),              // td1 hi=0 | tile_dim0 = 1024
              rows,                            // tile_dim1 | tile_dim2=0
              1024,                            // tensor_dim0_stride [31:0]
              0, 0};                           // stride hi | dim1_stride = 0
  i32x4 z4 = {0, 0, 0, 0};
  i32x8 z8 = {0, 0, 0, 0, 0, 0, 0, 0};
  __builtin_amdgcn_tensor_load_to_lds(g0, g1, z4, z4, z8, 0);
}
#else
#define HAVE_TDM 0
#endif

// ---------------------------------------------------------------------------
// Persistent GRU scan: grid = 32 blocks x 64 threads (2 waves).
// Wave gw owns h-indices [gw*16, gw*16+16) and all three gates for them.
// w_hh slice lives in LDS as [gate][32 rows][1024] bf16 = 192KB per WGP,
// staged by 3 TDM tensor loads (each gate's chunk is contiguous in global
// AND in LDS). h_{t-1} is broadcast into all 16 WMMA A-rows, so row 0 of
// each f32 accumulator is the matvec. One grid barrier per step.
// ---------------------------------------------------------------------------
__global__ void __launch_bounds__(64)
gru_scan(const float* __restrict__ gx,      // [T, 3H] precomputed x-gates+b_ih
         const uint16_t* __restrict__ whh,  // [3H, H] bf16
         const float* __restrict__ bhh,     // [3H]
         float* __restrict__ hbf,           // [2, H] f32 double buffer
         uint16_t* __restrict__ hbb,        // [2, H] bf16 double buffer
         float* __restrict__ outf,          // [T, H] or null
         uint16_t* __restrict__ outb,       // [T, H] bf16 or null
         int* __restrict__ bar,             // {count, generation}
         int T) {
  extern __shared__ uint16_t smem[];  // [3 gates][32 rows][1024]
  const int H = H_DIM;
  const int lane = threadIdx.x & 31;
  const int wave = threadIdx.x >> 5;
  const int nblocks = gridDim.x;

#if HAVE_TDM
  if (wave == 0) {
    const unsigned lds0 = (unsigned)(uintptr_t)&smem[0];
#pragma unroll
    for (int g = 0; g < 3; ++g) {
      const unsigned long long ga =
          (unsigned long long)(uintptr_t)whh +
          ((unsigned long long)(g * H + blockIdx.x * 32) * H) * 2ull;
      tdm_load_2d(lds0 + (unsigned)g * 65536u, ga, 32);
    }
    __builtin_amdgcn_s_wait_tensorcnt(0);
  }
  __syncthreads();
#else
  for (int r = 0; r < 96; ++r) {  // r = g*32 + row
    const int g = r >> 5, row = r & 31;
    const uint4* src =
        (const uint4*)(whh + (long)(g * H + blockIdx.x * 32 + row) * H);
    uint4* dst = (uint4*)(smem + (long)r * H);
    for (int q = threadIdx.x; q < H / 8; q += blockDim.x) dst[q] = src[q];
  }
  __syncthreads();
#endif

  const int gw = blockIdx.x * 2 + wave;
  const int j0 = gw * 16;
  const int jl = j0 + (lane & 15);
  const int kb = (lane & 16) ? 8 : 0;
  const float bhr = bhh[jl];
  const float bhz = bhh[H + jl];
  const float bhn = bhh[2 * H + jl];

  // LDS row for gate g, this wave's columns: g*32 + wave*16 + (lane&15)
  const uint16_t* b0p = smem + (long)(0 * 32 + wave * 16 + (lane & 15)) * H + kb;
  const uint16_t* b1p = smem + (long)(1 * 32 + wave * 16 + (lane & 15)) * H + kb;
  const uint16_t* b2p = smem + (long)(2 * 32 + wave * 16 + (lane & 15)) * H + kb;

  for (int t = 0; t < T; ++t) {
    const int p = t & 1;
    const uint16_t* hb = hbb + p * H + kb;
    v8f a0 = {}, a1 = {}, a2 = {};
#pragma unroll 2
    for (int kc = 0; kc < 32; ++kc) {
      Frag32 a, b0, b1, b2;
      a.u[0] = *(const uint4*)(hb + kc * 32);
      a.u[1] = *(const uint4*)(hb + kc * 32 + 16);
      b0.u[0] = *(const uint4*)(b0p + kc * 32);
      b0.u[1] = *(const uint4*)(b0p + kc * 32 + 16);
      b1.u[0] = *(const uint4*)(b1p + kc * 32);
      b1.u[1] = *(const uint4*)(b1p + kc * 32 + 16);
      b2.u[0] = *(const uint4*)(b2p + kc * 32);
      b2.u[1] = *(const uint4*)(b2p + kc * 32 + 16);
      a0 = WMMA_BF16(a.v, b0.v, a0);
      a1 = WMMA_BF16(a.v, b1.v, a1);
      a2 = WMMA_BF16(a.v, b2.v, a2);
    }
    if (lane < 16) {
      const float* gxt = gx + (long)t * G_DIM;
      const float r = sigmoidf_(gxt[jl] + a0[0] + bhr);
      const float z = sigmoidf_(gxt[H + jl] + a1[0] + bhz);
      const float nn = tanhf(gxt[2 * H + jl] + r * (a2[0] + bhn));
      const float hp = hbf[p * H + jl];
      const float hnew = (1.0f - z) * nn + z * hp;
      hbf[(p ^ 1) * H + jl] = hnew;
      hbb[(p ^ 1) * H + jl] = f2bf(hnew);
      if (outf) outf[(long)t * H + jl] = hnew;
      if (outb) outb[(long)t * H + jl] = f2bf(hnew);
    }
    // ---- grid-wide barrier (arrive + generation spin) ----
    __threadfence();
    __syncthreads();
    if (threadIdx.x == 0) {
      const int g = __hip_atomic_load(&bar[1], __ATOMIC_RELAXED,
                                      __HIP_MEMORY_SCOPE_AGENT);
      const int arr = __hip_atomic_fetch_add(&bar[0], 1, __ATOMIC_ACQ_REL,
                                             __HIP_MEMORY_SCOPE_AGENT);
      if (arr == nblocks - 1) {
        __hip_atomic_store(&bar[0], 0, __ATOMIC_RELAXED,
                           __HIP_MEMORY_SCOPE_AGENT);
        __hip_atomic_fetch_add(&bar[1], 1, __ATOMIC_RELEASE,
                               __HIP_MEMORY_SCOPE_AGENT);
      } else {
        while (__hip_atomic_load(&bar[1], __ATOMIC_ACQUIRE,
                                 __HIP_MEMORY_SCOPE_AGENT) == g) {
          __builtin_amdgcn_s_sleep(2);
        }
      }
    }
    __syncthreads();
  }
}

// ---------------------------------------------------------------------------
// BN column stats folded into the FC head:
//   g[j] = gamma[j]*rsqrt(var_j+eps)*fcw[j];  cpart[j] = beta[j]*fcw[j]-mu_j*g[j]
// ---------------------------------------------------------------------------
__global__ void __launch_bounds__(256)
bn_stats(const float* __restrict__ h2, const float* __restrict__ gamma,
         const float* __restrict__ beta, const float* __restrict__ fcw,
         float* __restrict__ gcoef, float* __restrict__ cpart, int T) {
  const int j = blockIdx.x;
  const int H = H_DIM;
  float s = 0.0f, ss = 0.0f;
  for (int t = threadIdx.x; t < T; t += blockDim.x) {
    const float v = h2[(long)t * H + j];
    s += v;
    ss += v * v;
  }
  __shared__ float rs[256], rss[256];
  rs[threadIdx.x] = s;
  rss[threadIdx.x] = ss;
  __syncthreads();
  for (int o = 128; o > 0; o >>= 1) {
    if (threadIdx.x < o) {
      rs[threadIdx.x] += rs[threadIdx.x + o];
      rss[threadIdx.x] += rss[threadIdx.x + o];
    }
    __syncthreads();
  }
  if (threadIdx.x == 0) {
    const float mu = rs[0] / (float)T;
    const float var = rss[0] / (float)T - mu * mu;
    const float rq = rsqrtf(var + 1e-5f);
    const float g = gamma[j] * rq * fcw[j];
    gcoef[j] = g;
    cpart[j] = beta[j] * fcw[j] - mu * g;
  }
}

__global__ void __launch_bounds__(256)
bn_c(const float* __restrict__ cpart, const float* __restrict__ fcb,
     float* __restrict__ cout) {
  __shared__ float r[256];
  float s = 0.0f;
  for (int j = threadIdx.x; j < H_DIM; j += 256) s += cpart[j];
  r[threadIdx.x] = s;
  __syncthreads();
  for (int o = 128; o > 0; o >>= 1) {
    if (threadIdx.x < o) r[threadIdx.x] += r[threadIdx.x + o];
    __syncthreads();
  }
  if (threadIdx.x == 0) cout[0] = r[0] + fcb[0];
}

// y[t] = h2[t] . g + c    (one wave per row)
__global__ void __launch_bounds__(256)
head(const float* __restrict__ h2, const float* __restrict__ g,
     const float* __restrict__ c, float* __restrict__ y, int T) {
  const int lane = threadIdx.x & 31;
  const int t = (int)(((long)blockIdx.x * blockDim.x + threadIdx.x) >> 5);
  if (t >= T) return;
  const float* row = h2 + (long)t * H_DIM;
  float s = 0.0f;
  for (int j = lane; j < H_DIM; j += 32) s += row[j] * g[j];
  for (int o = 16; o > 0; o >>= 1) s += __shfl_down(s, o, 32);
  if (lane == 0) y[t] = s + c[0];
}

// ---------------------------------------------------------------------------
extern "C" void kernel_launch(void* const* d_in, const int* in_sizes, int n_in,
                              void* d_out, int out_size, void* d_ws,
                              size_t ws_size, hipStream_t stream) {
  (void)in_sizes; (void)n_in; (void)out_size; (void)ws_size;
  const float* x     = (const float*)d_in[0];
  const float* wih0  = (const float*)d_in[1];
  const float* whh0  = (const float*)d_in[2];
  const float* bih0  = (const float*)d_in[3];
  const float* bhh0  = (const float*)d_in[4];
  const float* wih1  = (const float*)d_in[5];
  const float* whh1  = (const float*)d_in[6];
  const float* bih1  = (const float*)d_in[7];
  const float* bhh1  = (const float*)d_in[8];
  const float* gamma = (const float*)d_in[9];
  const float* beta  = (const float*)d_in[10];
  const float* fcw   = (const float*)d_in[11];
  const float* fcb   = (const float*)d_in[12];
  float* y = (float*)d_out;

  char* ws = (char*)d_ws;
  size_t off = 0;
  auto take = [&](size_t bytes) -> char* {
    char* p = ws + off;
    off = (off + bytes + 255) & ~(size_t)255;
    return p;
  };
  uint16_t* xb    = (uint16_t*)take((size_t)T_LEN * IN_DIM * 2);
  uint16_t* wih0b = (uint16_t*)take((size_t)G_DIM * IN_DIM * 2);
  uint16_t* whh0b = (uint16_t*)take((size_t)G_DIM * H_DIM * 2);
  uint16_t* wih1b = (uint16_t*)take((size_t)G_DIM * H_DIM * 2);
  uint16_t* whh1b = (uint16_t*)take((size_t)G_DIM * H_DIM * 2);
  uint16_t* h1b   = (uint16_t*)take((size_t)T_LEN * H_DIM * 2);
  float*    gxbuf = (float*)take((size_t)T_LEN * G_DIM * 4);
  float*    h2    = (float*)take((size_t)T_LEN * H_DIM * 4);
  float*    hbf   = (float*)take(2 * H_DIM * 4);
  uint16_t* hbb   = (uint16_t*)take(2 * H_DIM * 2);
  int*      bar   = (int*)take(256);
  float*    gco   = (float*)take(H_DIM * 4);
  float*    cpart = (float*)take(H_DIM * 4);
  float*    csc   = (float*)take(256);

  (void)hipFuncSetAttribute(reinterpret_cast<const void*>(gru_scan),
                            hipFuncAttributeMaxDynamicSharedMemorySize, 196608);

  auto cvt = [&](const float* s, uint16_t* d, long n) {
    f32_to_bf16<<<dim3((unsigned)((n + 255) / 256)), dim3(256), 0, stream>>>(s, d, n);
  };
  cvt(x, xb, (long)T_LEN * IN_DIM);
  cvt(wih0, wih0b, (long)G_DIM * IN_DIM);
  cvt(whh0, whh0b, (long)G_DIM * H_DIM);
  cvt(wih1, wih1b, (long)G_DIM * H_DIM);
  cvt(whh1, whh1b, (long)G_DIM * H_DIM);

  const long tiles = (long)(T_LEN / 32) * (G_DIM / 32);

  // Layer 0: gx = x @ w_ih0^T + b_ih0; then scan -> h1 (bf16)
  gemm_bias_bf16<<<dim3((unsigned)((tiles + 3) / 4)), dim3(128), 0, stream>>>(
      xb, wih0b, bih0, gxbuf, T_LEN, G_DIM, IN_DIM);
  scan_init<<<dim3(8), dim3(256), 0, stream>>>(bar, hbf, hbb);
  gru_scan<<<dim3(32), dim3(64), 196608, stream>>>(
      gxbuf, whh0b, bhh0, hbf, hbb, nullptr, h1b, bar, T_LEN);

  // Layer 1: gx = h1 @ w_ih1^T + b_ih1; then scan -> h2 (f32)
  gemm_bias_bf16<<<dim3((unsigned)((tiles + 3) / 4)), dim3(128), 0, stream>>>(
      h1b, wih1b, bih1, gxbuf, T_LEN, G_DIM, H_DIM);
  scan_init<<<dim3(8), dim3(256), 0, stream>>>(bar, hbf, hbb);
  gru_scan<<<dim3(32), dim3(64), 196608, stream>>>(
      gxbuf, whh1b, bhh1, hbf, hbb, h2, nullptr, bar, T_LEN);

  // BatchNorm (training stats) folded into 1-output FC head.
  bn_stats<<<dim3(H_DIM), dim3(256), 0, stream>>>(h2, gamma, beta, fcw, gco,
                                                  cpart, T_LEN);
  bn_c<<<dim3(1), dim3(256), 0, stream>>>(cpart, fcb, csc);
  head<<<dim3((T_LEN * 32) / 256), dim3(256), 0, stream>>>(h2, gco, csc, y,
                                                           T_LEN);
}
